// OriginalSpatialAttention_80693845557762
// MI455X (gfx1250) — compile-verified, matrix operand order checked by reading, not run on hardware
//
#include <hip/hip_runtime.h>
#include <hip/hip_bf16.h>

typedef __attribute__((ext_vector_type(16))) _Float16 v16h;
typedef __attribute__((ext_vector_type(8)))  _Float16 v8h;
typedef __attribute__((ext_vector_type(8)))  float    v8f;

#define HEADS      6
#define HD         32
#define C_DIM      192
#define NWIN       128      // tokens per window (8x16)
#define TOTAL_ROWS 65536    // B * H * W = 4 * 128 * 128
#define QKV_COLS   576
#define SCALE_F    0.17677669529663687f   // 32^-0.5

// ---------------------------------------------------------------------------
// CDNA5 async global->LDS helpers (ASYNCcnt path, CDNA5 ISA)
// ---------------------------------------------------------------------------
static __device__ __forceinline__ void async_copy_b128(unsigned lds_byte_addr,
                                                       const void* sbase,
                                                       int byte_off) {
  // GVS mode: global addr = SGPR64 base + VGPR32 offset; VDST = LDS byte addr
  asm volatile("global_load_async_to_lds_b128 %0, %1, %2"
               :
               : "v"(lds_byte_addr), "v"(byte_off), "s"(sbase)
               : "memory");
}
static __device__ __forceinline__ void wait_async0() {
  asm volatile("s_wait_asynccnt 0x0" ::: "memory");
}

// ---------------------------------------------------------------------------
// K0: convert x fp32 -> f16 once (removes cvt chains + halves GEMM A traffic)
// ---------------------------------------------------------------------------
__global__ void __launch_bounds__(256) k_cvt_x(
    const float* __restrict__ x, _Float16* __restrict__ x16) {
  size_t i = (size_t)blockIdx.x * 256 + threadIdx.x;   // 8 elements per thread
  const float4* p = reinterpret_cast<const float4*>(x) + 2 * i;
  float4 a = p[0], b = p[1];
  v8h o;
  o[0] = (_Float16)a.x; o[1] = (_Float16)a.y; o[2] = (_Float16)a.z; o[3] = (_Float16)a.w;
  o[4] = (_Float16)b.x; o[5] = (_Float16)b.y; o[6] = (_Float16)b.z; o[7] = (_Float16)b.w;
  *reinterpret_cast<v8h*>(x16 + 8 * i) = o;
}

// ---------------------------------------------------------------------------
// K1: convert qkv / proj weights fp32 -> f16 scratch
// ---------------------------------------------------------------------------
__global__ void __launch_bounds__(256) k_cvt_weights(
    const float* __restrict__ wqkv, const float* __restrict__ wproj,
    _Float16* __restrict__ wqkv16, _Float16* __restrict__ wproj16) {
  int i = blockIdx.x * 256 + threadIdx.x;
  if (i < QKV_COLS * C_DIM) wqkv16[i] = (_Float16)wqkv[i];
  if (i < C_DIM * C_DIM)    wproj16[i] = (_Float16)wproj[i];
}

// ---------------------------------------------------------------------------
// K2: dynamic position bias MLP: 465 rel positions, pd=48, 3 LN+ReLU+FC layers
// ---------------------------------------------------------------------------
static __device__ __forceinline__ void ln_relu48(float* t, const float* g, const float* b) {
  float m = 0.f;
#pragma unroll
  for (int j = 0; j < 48; ++j) m += t[j];
  m *= (1.0f / 48.0f);
  float v = 0.f;
#pragma unroll
  for (int j = 0; j < 48; ++j) { float d = t[j] - m; v += d * d; }
  v *= (1.0f / 48.0f);
  float r = rsqrtf(v + 1e-5f);
#pragma unroll
  for (int j = 0; j < 48; ++j) {
    float y = (t[j] - m) * r * g[j] + b[j];
    t[j] = fmaxf(y, 0.f);
  }
}

__global__ void __launch_bounds__(512) k_pos_mlp(
    const float* __restrict__ ppw, const float* __restrict__ ppb,
    const float* __restrict__ g1, const float* __restrict__ be1,
    const float* __restrict__ w1, const float* __restrict__ b1,
    const float* __restrict__ g2, const float* __restrict__ be2,
    const float* __restrict__ w2, const float* __restrict__ b2,
    const float* __restrict__ g3, const float* __restrict__ be3,
    const float* __restrict__ w3, const float* __restrict__ b3,
    float* __restrict__ pout) {
  int i = blockIdx.x * blockDim.x + threadIdx.x;
  if (i >= 465) return;                    // (2*8-1)*(2*16-1)
  float bh = (float)((i / 31) - 7);
  float bw = (float)((i % 31) - 15);
  float t[48], u[48];
#pragma unroll
  for (int j = 0; j < 48; ++j) t[j] = ppw[2*j] * bh + ppw[2*j+1] * bw + ppb[j];
  ln_relu48(t, g1, be1);
  for (int j = 0; j < 48; ++j) {
    float s = b1[j];
#pragma unroll
    for (int k = 0; k < 48; ++k) s += t[k] * w1[j*48 + k];
    u[j] = s;
  }
  ln_relu48(u, g2, be2);
  for (int j = 0; j < 48; ++j) {
    float s = b2[j];
#pragma unroll
    for (int k = 0; k < 48; ++k) s += u[k] * w2[j*48 + k];
    t[j] = s;
  }
  ln_relu48(t, g3, be3);
#pragma unroll
  for (int h = 0; h < HEADS; ++h) {
    float s = b3[h];
#pragma unroll
    for (int k = 0; k < 48; ++k) s += t[k] * w3[h*48 + k];
    pout[i * HEADS + h] = s;
  }
}

// ---------------------------------------------------------------------------
// K3: gather MLP output through relative-position index into (6,128,128)
// ---------------------------------------------------------------------------
__global__ void __launch_bounds__(256) k_rpb_gather(
    const float* __restrict__ p, float* __restrict__ rpb) {
  int id = blockIdx.x * 256 + threadIdx.x;     // n*128 + m
  if (id >= NWIN * NWIN) return;
  int n = id >> 7, m = id & 127;
  int idx = ((n >> 4) - (m >> 4) + 7) * 31 + ((n & 15) - (m & 15) + 15);
#pragma unroll
  for (int h = 0; h < HEADS; ++h)
    rpb[h * (NWIN * NWIN) + id] = p[idx * HEADS + h];
}

// ---------------------------------------------------------------------------
// K4: QKV GEMM (65536x192 @ 192x576) via WMMA f16. Weight B-tiles staged in
// LDS by async global->LDS copies (shared by all 8 waves). Fused bias + scale
// + im2win scatter: q,k (wh,n,d); v transposed (wh,d,n).
// block = 8 waves = 128 rows x 64 cols; wave = 16 rows x 64 cols (4 tiles)
// ---------------------------------------------------------------------------
__global__ void __launch_bounds__(256) k_qkv_gemm(
    const _Float16* __restrict__ x16, const _Float16* __restrict__ w16,
    const float* __restrict__ bias,
    _Float16* __restrict__ q16, _Float16* __restrict__ k16,
    _Float16* __restrict__ v16) {
  __shared__ _Float16 bsm[64 * 32];            // 4 KB weight tile
  const int tid  = threadIdx.x;
  const int lane = tid & 31;
  const int wave = tid >> 5;
  const int lm = lane & 15;
  const int hi = lane >> 4;
  const int rowBase = blockIdx.y * 128 + wave * 16;
  const int colBase = blockIdx.x * 64;
  const int srow = tid >> 2, sseg = tid & 3;   // each thread stages one b128
  const unsigned lds_dst =
      (unsigned)(size_t)&bsm[0] + (unsigned)(srow * 64 + sseg * 16);

  v8f c[4] = {};
  for (int k0 = 0; k0 < C_DIM; k0 += 32) {
    int goff = ((colBase + srow) * C_DIM + k0) * 2 + sseg * 16;
    async_copy_b128(lds_dst, w16, goff);
    wait_async0();
    __syncthreads();

    v16h a = *reinterpret_cast<const v16h*>(
        x16 + (size_t)(rowBase + lm) * C_DIM + k0 + hi * 16);
#pragma unroll
    for (int nt = 0; nt < 4; ++nt) {
      v16h b = *reinterpret_cast<const v16h*>(&bsm[(nt * 16 + lm) * 32 + hi * 16]);
      c[nt] = __builtin_amdgcn_wmma_f32_16x16x32_f16(
          false, a, false, b, (short)0, c[nt], false, false);
    }
    __syncthreads();
  }

#pragma unroll
  for (int nt = 0; nt < 4; ++nt) {
    int col = colBase + nt * 16 + lm;
    int part = col / C_DIM;                   // 0=q 1=k 2=v
    int cc = col - part * C_DIM;
    int head = cc >> 5, d = cc & 31;
    float bv = bias[col];
#pragma unroll
    for (int i = 0; i < 8; ++i) {
      int row = rowBase + i + 8 * hi;
      int bidx = row >> 14;                   // / 16384
      int s = row & 16383;
      int y = s >> 7, xx = s & 127;
      int wy = y >> 3, iy = y & 7;
      int wx = xx >> 4, ix = xx & 15;
      int widx = bidx * 128 + wy * 8 + wx;
      int ntok = iy * 16 + ix;
      size_t base = (size_t)(widx * HEADS + head) * (NWIN * HD);
      float val = c[nt][i] + bv;
      if (part == 0)      q16[base + ntok * HD + d] = (_Float16)(val * SCALE_F);
      else if (part == 1) k16[base + ntok * HD + d] = (_Float16)val;
      else                v16[base + d * NWIN + ntok] = (_Float16)val; // transposed
    }
  }
}

// ---------------------------------------------------------------------------
// K5: fused attention per (window, head). 8 waves; wave w owns rows [16w,16w+16).
// QK^T: 8 WMMA (K=32). + rpb + mask, shuffle softmax, f16 repack via LDS,
// PV: 8 WMMA (K=128 over 4 slices, N=32 over 2 tiles). Writes (B,L,C) f16.
// ---------------------------------------------------------------------------
__global__ void __launch_bounds__(256) k_attn(
    const _Float16* __restrict__ q16, const _Float16* __restrict__ k16,
    const _Float16* __restrict__ v16, const float* __restrict__ rpb,
    const float* __restrict__ mask, _Float16* __restrict__ out16) {
  __shared__ _Float16 sm[8 * 16 * NWIN];     // 32 KB: 16x128 f16 per wave
  const int lane = threadIdx.x & 31;
  const int wave = threadIdx.x >> 5;
  const int lm = lane & 15;
  const int hi = lane >> 4;

  const int wh = blockIdx.x;                 // widx*6 + head
  const int widx = wh / HEADS;
  const int head = wh - widx * HEADS;
  const _Float16* qb = q16 + (size_t)wh * (NWIN * HD);
  const _Float16* kb = k16 + (size_t)wh * (NWIN * HD);
  const _Float16* vb = v16 + (size_t)wh * (NWIN * HD);
  const int r0 = wave * 16;

  // ---- S = (q*scale) @ k^T : one WMMA per 16x16 tile (K = 32 = head_dim)
  v16h a = *reinterpret_cast<const v16h*>(qb + (r0 + lm) * HD + hi * 16);
  v8f c[8] = {};
#pragma unroll
  for (int t = 0; t < 8; ++t) {
    v16h b = *reinterpret_cast<const v16h*>(kb + (t * 16 + lm) * HD + hi * 16);
    c[t] = __builtin_amdgcn_wmma_f32_16x16x32_f16(
        false, a, false, b, (short)0, c[t], false, false);
  }

  // ---- + relative position bias + mask
  const float* rp = rpb + (size_t)head * (NWIN * NWIN);
  const float* mp = mask + (size_t)(widx & 127) * (NWIN * NWIN);
#pragma unroll
  for (int t = 0; t < 8; ++t)
#pragma unroll
    for (int i = 0; i < 8; ++i) {
      int n = r0 + i + 8 * hi;
      int m = t * 16 + lm;
      c[t][i] += rp[n * NWIN + m] + mp[n * NWIN + m];
    }

  // ---- softmax along m (rows live in 16-lane halves: 4 xor-shuffle steps)
#pragma unroll
  for (int i = 0; i < 8; ++i) {
    float mv = -1e30f;
#pragma unroll
    for (int t = 0; t < 8; ++t) mv = fmaxf(mv, c[t][i]);
    mv = fmaxf(mv, __shfl_xor(mv, 8));
    mv = fmaxf(mv, __shfl_xor(mv, 4));
    mv = fmaxf(mv, __shfl_xor(mv, 2));
    mv = fmaxf(mv, __shfl_xor(mv, 1));
    float s = 0.f;
#pragma unroll
    for (int t = 0; t < 8; ++t) {
      float e = __expf(c[t][i] - mv);
      c[t][i] = e;
      s += e;
    }
    s += __shfl_xor(s, 8);
    s += __shfl_xor(s, 4);
    s += __shfl_xor(s, 2);
    s += __shfl_xor(s, 1);
    float inv = 1.0f / s;
#pragma unroll
    for (int t = 0; t < 8; ++t) c[t][i] *= inv;
  }

  // ---- repack C-layout f32 -> A-layout f16 through LDS (same-wave, in-order)
  _Float16* ws = sm + wave * (16 * NWIN);
#pragma unroll
  for (int t = 0; t < 8; ++t)
#pragma unroll
    for (int i = 0; i < 8; ++i)
      ws[(i + 8 * hi) * NWIN + t * 16 + lm] = (_Float16)c[t][i];

  // ---- O = P @ v : K=128 in 4 slices, N=32 in 2 tiles (v stored transposed)
  v8f o[2] = {};
#pragma unroll
  for (int ks = 0; ks < 4; ++ks) {
    v16h aa = *reinterpret_cast<const v16h*>(ws + lm * NWIN + ks * 32 + hi * 16);
#pragma unroll
    for (int nt = 0; nt < 2; ++nt) {
      v16h bb = *reinterpret_cast<const v16h*>(vb + (nt * 16 + lm) * NWIN + ks * 32 + hi * 16);
      o[nt] = __builtin_amdgcn_wmma_f32_16x16x32_f16(
          false, aa, false, bb, (short)0, o[nt], false, false);
    }
  }

  // ---- windows2img scatter into (B, L, C) f16 for the proj GEMM
  const int bidx = widx >> 7;
  const int wib = widx & 127;
  const int wy = wib >> 3, wx = wib & 7;
#pragma unroll
  for (int nt = 0; nt < 2; ++nt)
#pragma unroll
    for (int i = 0; i < 8; ++i) {
      int ntok = r0 + i + 8 * hi;
      int iy = ntok >> 4, ix = ntok & 15;
      int grow = bidx * 16384 + (wy * 8 + iy) * 128 + wx * 16 + ix;
      int d = nt * 16 + lm;
      out16[(size_t)grow * C_DIM + head * HD + d] = (_Float16)o[nt][i];
    }
}

// ---------------------------------------------------------------------------
// K6: output projection (65536x192 @ 192x192) + bias -> fp32 d_out
// weights staged in LDS via async copies, same scheme as K4
// ---------------------------------------------------------------------------
__global__ void __launch_bounds__(256) k_proj_gemm(
    const _Float16* __restrict__ in16, const _Float16* __restrict__ w16,
    const float* __restrict__ bias, float* __restrict__ out) {
  __shared__ _Float16 bsm[64 * 32];
  const int tid  = threadIdx.x;
  const int lane = tid & 31;
  const int wave = tid >> 5;
  const int lm = lane & 15;
  const int hi = lane >> 4;
  const int rowBase = blockIdx.y * 128 + wave * 16;
  const int colBase = blockIdx.x * 64;
  const int srow = tid >> 2, sseg = tid & 3;
  const unsigned lds_dst =
      (unsigned)(size_t)&bsm[0] + (unsigned)(srow * 64 + sseg * 16);

  v8f c[4] = {};
  for (int k0 = 0; k0 < C_DIM; k0 += 32) {
    int goff = ((colBase + srow) * C_DIM + k0) * 2 + sseg * 16;
    async_copy_b128(lds_dst, w16, goff);
    wait_async0();
    __syncthreads();

    v16h a = *reinterpret_cast<const v16h*>(
        in16 + (size_t)(rowBase + lm) * C_DIM + k0 + hi * 16);
#pragma unroll
    for (int nt = 0; nt < 4; ++nt) {
      v16h b = *reinterpret_cast<const v16h*>(&bsm[(nt * 16 + lm) * 32 + hi * 16]);
      c[nt] = __builtin_amdgcn_wmma_f32_16x16x32_f16(
          false, a, false, b, (short)0, c[nt], false, false);
    }
    __syncthreads();
  }

#pragma unroll
  for (int nt = 0; nt < 4; ++nt) {
    int col = colBase + nt * 16 + lm;
    float bv = bias[col];
#pragma unroll
    for (int i = 0; i < 8; ++i) {
      int row = rowBase + i + 8 * hi;
      out[(size_t)row * C_DIM + col] = c[nt][i] + bv;
    }
  }
}

// ---------------------------------------------------------------------------
// launch
// ---------------------------------------------------------------------------
extern "C" void kernel_launch(void* const* d_in, const int* in_sizes, int n_in,
                              void* d_out, int out_size, void* d_ws, size_t ws_size,
                              hipStream_t stream) {
  (void)in_sizes; (void)n_in; (void)out_size; (void)ws_size;
  const float* x          = (const float*)d_in[0];
  const float* mask       = (const float*)d_in[1];
  const float* qkv_w      = (const float*)d_in[2];
  const float* qkv_b      = (const float*)d_in[3];
  const float* proj_w     = (const float*)d_in[4];
  const float* proj_b     = (const float*)d_in[5];
  const float* pos_proj_w = (const float*)d_in[6];
  const float* pos_proj_b = (const float*)d_in[7];
  const float* ln1_g = (const float*)d_in[8];
  const float* ln1_b = (const float*)d_in[9];
  const float* w1    = (const float*)d_in[10];
  const float* b1    = (const float*)d_in[11];
  const float* ln2_g = (const float*)d_in[12];
  const float* ln2_b = (const float*)d_in[13];
  const float* w2    = (const float*)d_in[14];
  const float* b2    = (const float*)d_in[15];
  const float* ln3_g = (const float*)d_in[16];
  const float* ln3_b = (const float*)d_in[17];
  const float* w3    = (const float*)d_in[18];
  const float* b3    = (const float*)d_in[19];
  float* out = (float*)d_out;

  // workspace carve-up (all chunk sizes are 256B multiples)
  const size_t qkv_half_bytes = (size_t)512 * HEADS * NWIN * HD * 2;  // 25.17 MB each
  char* ws = (char*)d_ws;
  _Float16* x16   = (_Float16*)ws;  ws += (size_t)TOTAL_ROWS * C_DIM * 2;
  _Float16* q16   = (_Float16*)ws;  ws += qkv_half_bytes;
  _Float16* k16   = (_Float16*)ws;  ws += qkv_half_bytes;
  _Float16* v16   = (_Float16*)ws;  ws += qkv_half_bytes;
  _Float16* ao16  = (_Float16*)ws;  ws += (size_t)TOTAL_ROWS * C_DIM * 2;
  _Float16* wq16  = (_Float16*)ws;  ws += (size_t)QKV_COLS * C_DIM * 2;
  _Float16* wp16  = (_Float16*)ws;  ws += (size_t)C_DIM * C_DIM * 2;
  float*    rpb   = (float*)ws;     ws += (size_t)HEADS * NWIN * NWIN * 4;
  float*    ptab  = (float*)ws;

  k_cvt_x<<<(TOTAL_ROWS * C_DIM / 8 + 255) / 256, 256, 0, stream>>>(x, x16);

  k_cvt_weights<<<(QKV_COLS * C_DIM + 255) / 256, 256, 0, stream>>>(
      qkv_w, proj_w, wq16, wp16);

  k_pos_mlp<<<1, 512, 0, stream>>>(
      pos_proj_w, pos_proj_b, ln1_g, ln1_b, w1, b1,
      ln2_g, ln2_b, w2, b2, ln3_g, ln3_b, w3, b3, ptab);

  k_rpb_gather<<<(NWIN * NWIN + 255) / 256, 256, 0, stream>>>(ptab, rpb);

  k_qkv_gemm<<<dim3(QKV_COLS / 64, TOTAL_ROWS / 128), 256, 0, stream>>>(
      x16, wq16, qkv_b, q16, k16, v16);

  k_attn<<<512 * HEADS, 256, 0, stream>>>(q16, k16, v16, rpb, mask, ao16);

  k_proj_gemm<<<dim3(C_DIM / 64, TOTAL_ROWS / 128), 256, 0, stream>>>(
      ao16, wp16, proj_b, out);
}
